// PathCorrectSampler_77086073029140
// MI455X (gfx1250) — matrix operand order repaired and checked by reading, not compiled, not run on hardware
//
#include <hip/hip_runtime.h>

// Problem constants (match reference)
#define BDIM 1024
#define DDIM 4096
#define MAXR 19

typedef __attribute__((ext_vector_type(16))) _Float16 v16h;
typedef __attribute__((ext_vector_type(8)))  _Float16 v8h;
typedef __attribute__((ext_vector_type(4)))  _Float16 v4h;
typedef __attribute__((ext_vector_type(8)))  float    v8f;
typedef __attribute__((ext_vector_type(4)))  unsigned int u32x4;
typedef __attribute__((ext_vector_type(8)))  int i32x8;
typedef __attribute__((ext_vector_type(4)))  int i32x4;

// ---------------------------------------------------------------------------
// Deterministic counter-based RNG (graph-replay safe)
// ---------------------------------------------------------------------------
__device__ __forceinline__ unsigned int ihash(unsigned int x) {
  x ^= x >> 16; x *= 0x7feb352du;
  x ^= x >> 15; x *= 0x846ca68bu;
  x ^= x >> 16;
  return x;
}
__device__ __forceinline__ float u01(unsigned int h) {
  return ((h >> 8) + 0.5f) * (1.0f / 16777216.0f);   // (0,1)
}

// ---------------------------------------------------------------------------
// f32 -> f16 pack
// ---------------------------------------------------------------------------
__global__ void pack_f32_to_f16(const float* __restrict__ src,
                                _Float16* __restrict__ dst, size_t n) {
  size_t i = (size_t)blockIdx.x * blockDim.x + threadIdx.x;
  size_t stride = (size_t)gridDim.x * blockDim.x;
  for (; i < n; i += stride) dst[i] = (_Float16)src[i];
}

// ---------------------------------------------------------------------------
// TDM: issue a 2D tile load Global->LDS via tensor descriptor (D#).
// Bit packing per CDNA5 ISA.
//   group0: [1:0]=count=1, [63:32]=lds_addr, [120:64]=global_addr, [127:126]=2
//   group1: [17:16]=data_size(1 -> 2 bytes), [79:48]=tensor_dim0,
//           [111:80]=tensor_dim1, [127:112]=tile_dim0, [143:128]=tile_dim1,
//           [207:160]=tensor_dim0_stride
// 2D tensor -> descriptor groups 2/3 (and trailing group) zero.
// This toolchain exposes the 6-arg builtin:
//   (u32x4 g0, i32x8 g1, i32x4 g2, i32x4 g3, i32x8 g4, i32 cpol)
// ---------------------------------------------------------------------------
__device__ __forceinline__ void tdm_load_2d(unsigned lds_off,
                                            const _Float16* gptr,
                                            unsigned tile_w, unsigned tile_h,
                                            unsigned tensor_w, unsigned tensor_h,
                                            unsigned row_stride_elems) {
  unsigned long long ga = (unsigned long long)(uintptr_t)gptr;
  u32x4 g0;
  g0[0] = 1u;                                            // count=1 (user D#)
  g0[1] = lds_off;                                       // LDS byte address
  g0[2] = (unsigned)(ga & 0xFFFFFFFFull);                // global addr [31:0]
  g0[3] = (unsigned)((ga >> 32) & 0x1FFFFFFull) | (2u << 30);  // [56:32]|type=2
  i32x8 g1;
  g1[0] = (int)(1u << 16);                               // data_size=1 (2B)
  g1[1] = (int)((tensor_w & 0xFFFFu) << 16);             // tensor_dim0 lo16
  g1[2] = (int)((tensor_w >> 16) | ((tensor_h & 0xFFFFu) << 16));
  g1[3] = (int)((tensor_h >> 16) | (tile_w << 16));      // tile_dim0
  g1[4] = (int)tile_h;                                   // tile_dim1 (dim2=0)
  g1[5] = (int)row_stride_elems;                         // dim0_stride lo32
  g1[6] = 0;                                             // dim0_stride hi16
  g1[7] = 0;
  i32x4 z4 = {0, 0, 0, 0};
  i32x8 z8 = {0, 0, 0, 0, 0, 0, 0, 0};
  __builtin_amdgcn_tensor_load_to_lds(g0, g1, z4, z4, z8, 0);
}

__device__ __forceinline__ unsigned lds_offset_of(const void* p) {
  // LDS aperture: addr[31:0] is the LDS byte offset (ISA §10.2 aperture calc)
  return (unsigned)(unsigned long long)(uintptr_t)p;
}

// ---------------------------------------------------------------------------
// WMMA GEMM: C[M,N] = A[M,K] * W[K,N] + bias[N]   (M=BDIM, N=K=DDIM)
// Block = 256 threads = 8 waves (2 M-waves x 4 N-waves).
// Block tile: 32(M) x 256(N), K-chunk 32. Each wave: 1 A-frag x 4 B-frags
// -> 4 v_wmma per K-chunk per wave.
// Tiles land in LDS via double-buffered TDM (tensor_load_to_lds), paced
// with s_wait_tensorcnt.
// ---------------------------------------------------------------------------
__global__ __launch_bounds__(256) void gemm_f16_wmma(
    const _Float16* __restrict__ A, const _Float16* __restrict__ Wt,
    const float* __restrict__ bias, float* __restrict__ C) {
  __shared__ __align__(16) _Float16 As[2][32][32];    // [buf][M][K]
  __shared__ __align__(16) _Float16 Bs[2][32][256];   // [buf][K][N] row-major

  const int tid  = threadIdx.x;
  const int lane = tid & 31;
  const int wave = tid >> 5;
  const int wm   = wave >> 2;       // 0..1  (M sub-tile)
  const int wn   = wave & 3;        // 0..3  (N quarter: 64 cols)
  const int mBase = blockIdx.y * 32;
  const int nBase = blockIdx.x * 256;

  const int la = lane & 15;
  const int lh = lane >> 4;
  // A fragment (16-bit A 16x32): lanes 0-15: K 0..7 & 16..23; lanes 16-31: +8
  const int aRow  = wm * 16 + la;
  const int kOffA = lh * 8;
  // B fragment: lane = column; lanes 0-15: K 0..7 & 16..23; lanes 16-31: +8
  const int kOffB = lh * 8;

  v8f acc[4] = {{}, {}, {}, {}};

  const int nIter = DDIM / 32;

  // prologue: stage K-chunk 0 into buffer 0
  if (tid == 0) {
    tdm_load_2d(lds_offset_of(&As[0][0][0]),
                A + (size_t)mBase * DDIM, 32, 32, DDIM, BDIM, DDIM);
    tdm_load_2d(lds_offset_of(&Bs[0][0][0]),
                Wt + nBase, 256, 32, DDIM, DDIM, DDIM);
  }

  for (int it = 0; it < nIter; ++it) {
    const int cur = it & 1;
    if (tid == 0) {
      if (it + 1 < nIter) {
        const int k1 = (it + 1) * 32;
        tdm_load_2d(lds_offset_of(&As[1 - cur][0][0]),
                    A + (size_t)mBase * DDIM + k1, 32, 32, DDIM, BDIM, DDIM);
        tdm_load_2d(lds_offset_of(&Bs[1 - cur][0][0]),
                    Wt + (size_t)k1 * DDIM + nBase, 256, 32, DDIM, DDIM, DDIM);
        __builtin_amdgcn_s_wait_tensorcnt(2);  // current pair done; next in flight
      } else {
        __builtin_amdgcn_s_wait_tensorcnt(0);
      }
    }
    __syncthreads();

    v8h a_lo = *(const v8h*)(&As[cur][aRow][kOffA]);
    v8h a_hi = *(const v8h*)(&As[cur][aRow][kOffA + 16]);
    v16h afrag = __builtin_shufflevector(a_lo, a_hi, 0, 1, 2, 3, 4, 5, 6, 7,
                                         8, 9, 10, 11, 12, 13, 14, 15);
#pragma unroll
    for (int f = 0; f < 4; ++f) {
      const int col = wn * 64 + f * 16 + la;
      v16h bfrag;
#pragma unroll
      for (int e = 0; e < 16; ++e) bfrag[e] = Bs[cur][kOffB + e + ((e >> 3) << 3)][col];
      acc[f] = __builtin_amdgcn_wmma_f32_16x16x32_f16(
          false, afrag, false, bfrag, (short)0, acc[f], false, false);
    }
    __syncthreads();
  }

  // C/D layout: lanes 0-15: N=lane, M=v; lanes 16-31: N=lane-16, M=v+8
  const int mrow = mBase + wm * 16 + lh * 8;
#pragma unroll
  for (int f = 0; f < 4; ++f) {
    const int col = nBase + wn * 64 + f * 16 + la;
    const float bb = bias[col];
#pragma unroll
    for (int v = 0; v < 8; ++v)
      C[(size_t)(mrow + v) * DDIM + col] = acc[f][v] + bb;
  }
}

// ---------------------------------------------------------------------------
// Per-row sampler: 19 Gumbel-max categorical steps, log_fwd accumulation.
// score_x = 0.5*(x.grad_x + x.b) since grad = Wx + b and W symmetric.
// ---------------------------------------------------------------------------
__global__ __launch_bounds__(256) void sampler_kernel(
    const float* __restrict__ x, const float* __restrict__ grad,
    const float* __restrict__ bias, _Float16* __restrict__ yh,
    int* __restrict__ idxs, int* __restrict__ rad_out,
    float* __restrict__ logfwd) {
  __shared__ float gx[DDIM];
  __shared__ float cx[DDIM];
  __shared__ float rv[256];
  __shared__ float r2[256];
  __shared__ int   ri[256];

  const int tid = threadIdx.x;
  const int row = blockIdx.x;
  const size_t base = (size_t)row * DDIM;

  float sx = 0.f;
  for (int i = tid; i < DDIM; i += 256) {
    float xv = x[base + i];
    float gv = grad[base + i];
    gx[i] = gv; cx[i] = xv;
    sx += xv * 0.5f * (gv + bias[i]);
  }
  rv[tid] = sx; __syncthreads();
  for (int s = 128; s > 0; s >>= 1) {
    if (tid < s) rv[tid] += rv[tid + s];
    __syncthreads();
  }
  const float score_x = rv[0];
  __syncthreads();

  const int radius = 1 + (int)(ihash(0xB5297A4Du ^ ((unsigned)row * 0x9E3779B1u)) % 19u);
  float lacc = 0.f;

  for (int t = 0; t < MAXR; ++t) {
    float best = -1e30f; int bi = 0; float maxl = -1e30f;
    for (int i = tid; i < DDIM; i += 256) {
      float logit = (1.f - 2.f * cx[i]) * gx[i] * 0.5f;
      unsigned int h = ihash(((unsigned)row * 0x9E3779B1u) ^
                             ((unsigned)t * 0x85EBCA77u + (unsigned)i * 0xC2B2AE3Du) ^
                             0x165667B1u);
      float g = -__logf(-__logf(u01(h)));
      float v = logit + g;
      if (v > best) { best = v; bi = i; }
      maxl = fmaxf(maxl, logit);
    }
    rv[tid] = best; ri[tid] = bi; r2[tid] = maxl;
    __syncthreads();
    for (int s = 128; s > 0; s >>= 1) {
      if (tid < s) {
        if (rv[tid + s] > rv[tid]) { rv[tid] = rv[tid + s]; ri[tid] = ri[tid + s]; }
        r2[tid] = fmaxf(r2[tid], r2[tid + s]);
      }
      __syncthreads();
    }
    const int sel = ri[0];
    const float mlog = r2[0];
    __syncthreads();

    float esum = 0.f;
    for (int i = tid; i < DDIM; i += 256) {
      float logit = (1.f - 2.f * cx[i]) * gx[i] * 0.5f;
      esum += __expf(logit - mlog);
    }
    rv[tid] = esum; __syncthreads();
    for (int s = 128; s > 0; s >>= 1) {
      if (tid < s) rv[tid] += rv[tid + s];
      __syncthreads();
    }
    const float lse = mlog + __logf(rv[0]);
    const float selLogit = (1.f - 2.f * cx[sel]) * gx[sel] * 0.5f;
    __syncthreads();

    if (t < radius) lacc += selLogit - lse;
    if (tid == 0) {
      idxs[row * MAXR + t] = sel;
      if (t < radius) cx[sel] = 1.f - cx[sel];
    }
    __syncthreads();
  }

  for (int i = tid; i < DDIM; i += 256) yh[base + i] = (_Float16)cx[i];
  if (tid == 0) { logfwd[row] = lacc + score_x; rad_out[row] = radius; }
}

// ---------------------------------------------------------------------------
// Backward log-prob + Metropolis accept; writes final output.
// traj_bwd step t uses state x_{t+1} with grad_y; x_19 == y.
// ---------------------------------------------------------------------------
__global__ __launch_bounds__(256) void bwd_kernel(
    const float* __restrict__ x, const float* __restrict__ grad,
    const float* __restrict__ bias, const int* __restrict__ idxs,
    const int* __restrict__ rad_in, const float* __restrict__ logfwd,
    float* __restrict__ out) {
  __shared__ float gy[DDIM];
  __shared__ float cx[DDIM];
  __shared__ float rv[256];
  __shared__ float r2[256];

  const int tid = threadIdx.x;
  const int row = blockIdx.x;
  const size_t base = (size_t)row * DDIM;

  for (int i = tid; i < DDIM; i += 256) {
    gy[i] = grad[base + i];
    cx[i] = x[base + i];
  }
  __syncthreads();
  const int radius = rad_in[row];
  float lacc = 0.f;

  for (int t = 0; t < MAXR; ++t) {
    const int sel = idxs[row * MAXR + t];
    if (tid == 0 && t < radius) cx[sel] = 1.f - cx[sel];   // advance to x_{t+1}
    __syncthreads();

    float maxl = -1e30f;
    for (int i = tid; i < DDIM; i += 256) {
      float logit = (1.f - 2.f * cx[i]) * gy[i] * 0.5f;
      maxl = fmaxf(maxl, logit);
    }
    r2[tid] = maxl; __syncthreads();
    for (int s = 128; s > 0; s >>= 1) {
      if (tid < s) r2[tid] = fmaxf(r2[tid], r2[tid + s]);
      __syncthreads();
    }
    const float mlog = r2[0]; __syncthreads();

    float esum = 0.f;
    for (int i = tid; i < DDIM; i += 256) {
      float logit = (1.f - 2.f * cx[i]) * gy[i] * 0.5f;
      esum += __expf(logit - mlog);
    }
    rv[tid] = esum; __syncthreads();
    for (int s = 128; s > 0; s >>= 1) {
      if (tid < s) rv[tid] += rv[tid + s];
      __syncthreads();
    }
    const float lse = mlog + __logf(rv[0]);
    const float selLogit = (1.f - 2.f * cx[sel]) * gy[sel] * 0.5f;
    if (t < radius) lacc += selLogit - lse;
    __syncthreads();
  }

  float sy = 0.f;
  for (int i = tid; i < DDIM; i += 256) sy += cx[i] * 0.5f * (gy[i] + bias[i]);
  rv[tid] = sy; __syncthreads();
  for (int s = 128; s > 0; s >>= 1) {
    if (tid < s) rv[tid] += rv[tid + s];
    __syncthreads();
  }
  const float score_y = rv[0];

  const float log_acc = (lacc + score_y) - logfwd[row];
  const float u = u01(ihash(0x2545F491u ^ ((unsigned)row * 0x9E3779B1u)));
  const bool accepted = (__expf(log_acc) >= u);

  for (int i = tid; i < DDIM; i += 256)
    out[base + i] = accepted ? cx[i] : x[base + i];
}

// ---------------------------------------------------------------------------
// Launch: pack W,x -> f16 | GEMM grad_x | sampler (y) | GEMM grad_y | accept
// ---------------------------------------------------------------------------
extern "C" void kernel_launch(void* const* d_in, const int* in_sizes, int n_in,
                              void* d_out, int out_size, void* d_ws, size_t ws_size,
                              hipStream_t stream) {
  (void)in_sizes; (void)n_in; (void)out_size; (void)ws_size;
  const float* x    = (const float*)d_in[0];
  const float* W    = (const float*)d_in[1];
  const float* bias = (const float*)d_in[2];
  float* out = (float*)d_out;

  char* ws = (char*)d_ws;
  _Float16* Wh = (_Float16*)ws;  ws += (size_t)DDIM * DDIM * sizeof(_Float16);
  _Float16* Xh = (_Float16*)ws;  ws += (size_t)BDIM * DDIM * sizeof(_Float16);
  float* gradb = (float*)ws;     ws += (size_t)BDIM * DDIM * sizeof(float);
  int* idxs    = (int*)ws;       ws += (size_t)BDIM * MAXR * sizeof(int);
  int* rad     = (int*)ws;       ws += (size_t)BDIM * sizeof(int);
  float* lfwd  = (float*)ws;     ws += (size_t)BDIM * sizeof(float);

  pack_f32_to_f16<<<4096, 256, 0, stream>>>(W, Wh, (size_t)DDIM * DDIM);
  pack_f32_to_f16<<<1024, 256, 0, stream>>>(x, Xh, (size_t)BDIM * DDIM);

  dim3 gg(DDIM / 256, BDIM / 32);
  gemm_f16_wmma<<<gg, 256, 0, stream>>>(Xh, Wh, bias, gradb);   // grad_x

  sampler_kernel<<<BDIM, 256, 0, stream>>>(x, gradb, bias, Xh, idxs, rad, lfwd);

  gemm_f16_wmma<<<gg, 256, 0, stream>>>(Xh, Wh, bias, gradb);   // grad_y (Xh==yh)

  bwd_kernel<<<BDIM, 256, 0, stream>>>(x, gradb, bias, idxs, rad, lfwd, out);
}
